// MoE_66434554135194
// MI455X (gfx1250) — matrix operand order repaired and checked by reading, not compiled, not run on hardware
//
#include <hip/hip_runtime.h>
#include <hip/hip_bf16.h>
#include <stdint.h>

// ---- problem constants (B=4,S=2048,H=1024,I=4096,E=8, top-2, cap 1.25) ----
constexpr int T_   = 8192;           // B*S tokens
constexpr int H_   = 1024;
constexpr int I_   = 4096;
constexpr int E_   = 8;
constexpr int CAP_ = 1280;           // ceil(T/E * 1.25)

// ---- WMMA / TDM types (CDNA5 gfx1250, wave32) ----
typedef __bf16        v16bf __attribute__((ext_vector_type(16)));
typedef float         v8f   __attribute__((ext_vector_type(8)));
typedef unsigned int  u32x4 __attribute__((ext_vector_type(4)));
typedef int           i32x8 __attribute__((ext_vector_type(8)));
typedef int           i32x4 __attribute__((ext_vector_type(4)));

// ---- GEMM tiling ----
constexpr int BM  = 64;              // block tile M
constexpr int BN  = 256;             // block tile N
constexpr int BK  = 32;              // K per WMMA step (16x16x32 bf16)
constexpr int LDT = BK + 2;          // LDS row stride (34 elems = 17 dwords, odd)

#if __has_builtin(__builtin_amdgcn_tensor_load_to_lds) && \
    __has_builtin(__builtin_amdgcn_s_wait_tensorcnt)
#define MOE_HAVE_TDM 1
#else
#define MOE_HAVE_TDM 0
#endif

#if MOE_HAVE_TDM
// Issue one TDM 2D tile load: 64 rows x 32 bf16, global row stride
// `stride_elems`, into LDS at byte offset `lds_off` with 1-DWORD padding
// after every 16 data DWORDs (matches LDS row stride LDT=34 elems).
// D# layout per CDNA5 ISA 8.3/8.4. EXEC ignored; issue from one wave only.
// amdgpu-toolchain (clang-23) 6-arg form:
//   (u32x4 g0, i32x8 g1, i32x4 g2, i32x4 g3, i32x8 g4, i32 cpol)
__device__ __forceinline__ void tdm_load_a_tile(const __bf16* gsrc,
                                                unsigned lds_off,
                                                unsigned stride_elems) {
  const unsigned long long ga = (unsigned long long)(uintptr_t)gsrc;
  u32x4 g0;
  g0[0] = 1u;                                           // count=1 (valid user D#)
  g0[1] = lds_off;                                      // lds_addr (bytes)
  g0[2] = (unsigned)(ga & 0xFFFFFFFFu);                 // global_addr[31:0]
  g0[3] = (unsigned)((ga >> 32) & 0x01FFFFFFu)          // global_addr[56:32]
          | (2u << 30);                                 // type=2 ("image")
  i32x8 g1;
  g1[0] = (int)((1u << 16)      // data_size = 2 bytes
              | (1u << 20)      // pad_enable
              | (3u << 22));    // pad_interval code 3 = 16 DWORDs; pad_amount 0 = 1 DWORD
  g1[1] = 0;                    // atomic_barrier_addr=0, tensor_dim0[15:0]=0
  g1[2] = 0x0010;               // tensor_dim0 = 1<<20 (hi), tensor_dim1[15:0]=0
  g1[3] = (32 << 16) | 0x0010;  // tile_dim0=32 elems; tensor_dim1 = 1<<20 (hi)
  g1[4] = 64;                   // tile_dim1 = 64 rows; tile_dim2 = 0 (2D)
  g1[5] = (int)stride_elems;    // tensor_dim0_stride[31:0] (data_size units)
  g1[6] = 0;                    // dim0_stride hi, dim1_stride lo
  g1[7] = 0;                    // dim1_stride hi
  i32x4 z4 = {0, 0, 0, 0};      // groups 2/3 unused (2D tile)
  i32x8 z8 = {0, 0, 0, 0, 0, 0, 0, 0};
  __builtin_amdgcn_tensor_load_to_lds(g0, g1, z4, z4, z8, 0);
}
#endif

// ===========================================================================
__global__ __launch_bounds__(256) void moe_zero(float* __restrict__ p) {
  int i = blockIdx.x * 256 + threadIdx.x;      // n4 = T*H/4 float4s
  float4 z; z.x = z.y = z.z = z.w = 0.f;
  reinterpret_cast<float4*>(p)[i] = z;
}

// ===========================================================================
// Router: logits = x @ Wr^T + b ; softmax(8) ; top-2 -> dense topk weights
__global__ __launch_bounds__(256) void moe_router(
    const float* __restrict__ x, const float* __restrict__ Wr,
    const float* __restrict__ br, float* __restrict__ topk_w) {
  const int t   = blockIdx.x;
  const int tid = threadIdx.x;
  const float4 xv = reinterpret_cast<const float4*>(x + (size_t)t * H_)[tid];
  float acc[E_];
#pragma unroll
  for (int e = 0; e < E_; ++e) {
    const float4 wv = reinterpret_cast<const float4*>(Wr + (size_t)e * H_)[tid];
    acc[e] = xv.x * wv.x + xv.y * wv.y + xv.z * wv.z + xv.w * wv.w;
  }
#pragma unroll
  for (int e = 0; e < E_; ++e) {
    float v = acc[e];
    for (int off = 16; off > 0; off >>= 1) v += __shfl_down(v, off);
    acc[e] = v;
  }
  __shared__ float wred[8][E_];
  __shared__ float lg[E_];
  const int wave = tid >> 5, lane = tid & 31;
  if (lane == 0) {
#pragma unroll
    for (int e = 0; e < E_; ++e) wred[wave][e] = acc[e];
  }
  __syncthreads();
  if (tid < E_) {
    float s = br[tid];
#pragma unroll
    for (int w = 0; w < 8; ++w) s += wred[w][tid];
    lg[tid] = s;
  }
  __syncthreads();
  if (tid == 0) {
    float l[E_], mx = -1e30f;
#pragma unroll
    for (int e = 0; e < E_; ++e) { l[e] = lg[e]; mx = l[e] > mx ? l[e] : mx; }
    float sum = 0.f;
#pragma unroll
    for (int e = 0; e < E_; ++e) { l[e] = __expf(l[e] - mx); sum += l[e]; }
    const float inv = 1.f / sum;
    int i1 = 0;
#pragma unroll
    for (int e = 1; e < E_; ++e) if (l[e] > l[i1]) i1 = e;   // first max on ties
    int i2 = -1;
#pragma unroll
    for (int e = 0; e < E_; ++e)
      if (e != i1 && (i2 < 0 || l[e] > l[i2])) i2 = e;       // first 2nd-max
#pragma unroll
    for (int e = 0; e < E_; ++e)
      topk_w[(size_t)t * E_ + e] = (e == i1 || e == i2) ? l[e] * inv : 0.f;
  }
}

// ===========================================================================
// Per-expert stable slot assignment (== stable argsort of binary mask).
__global__ __launch_bounds__(1024) void moe_assign(
    const float* __restrict__ topk_w, int* __restrict__ slot_token,
    float* __restrict__ slot_weight) {
  const int e   = blockIdx.x;
  const int tid = threadIdx.x;
  for (int i = tid; i < CAP_; i += 1024) {
    slot_token[e * CAP_ + i]  = -1;
    slot_weight[e * CAP_ + i] = 0.f;
  }
  __shared__ int sc[1024];
  __shared__ int sbase;
  if (tid == 0) sbase = 0;
  __syncthreads();
  for (int c = 0; c < T_ / 1024; ++c) {
    const int t = c * 1024 + tid;
    const float w = topk_w[(size_t)t * E_ + e];
    const int m = (w > 0.f) ? 1 : 0;
    sc[tid] = m;
    __syncthreads();
    for (int off = 1; off < 1024; off <<= 1) {      // inclusive scan
      int v = sc[tid];
      int add = (tid >= off) ? sc[tid - off] : 0;
      __syncthreads();
      sc[tid] = v + add;
      __syncthreads();
    }
    const int incl  = sc[tid];
    const int total = sc[1023];
    const int slot  = sbase + incl - m;
    if (m && slot < CAP_) {
      slot_token[e * CAP_ + slot]  = t;
      slot_weight[e * CAP_ + slot] = w;
    }
    __syncthreads();
    if (tid == 0) sbase += total;
    __syncthreads();
  }
}

// ===========================================================================
// Gather tokens -> bf16 A[E][CAP][H] (zeros for unfilled slots)
__global__ __launch_bounds__(256) void moe_gather(
    const float* __restrict__ x, const int* __restrict__ slot_token,
    __bf16* __restrict__ A) {
  const int slot = blockIdx.x, e = blockIdx.y, tid = threadIdx.x;
  const int token = slot_token[e * CAP_ + slot];
  __bf16* dst = A + ((size_t)e * CAP_ + slot) * H_ + tid * 4;
  if (token >= 0) {
    const float4 v = *reinterpret_cast<const float4*>(x + (size_t)token * H_ + tid * 4);
    dst[0] = (__bf16)v.x; dst[1] = (__bf16)v.y; dst[2] = (__bf16)v.z; dst[3] = (__bf16)v.w;
  } else {
    dst[0] = (__bf16)0.f; dst[1] = (__bf16)0.f; dst[2] = (__bf16)0.f; dst[3] = (__bf16)0.f;
  }
}

// ===========================================================================
// WMMA fragment assembly + 8x v_wmma_f32_16x16x32_bf16 per wave per K-step.
// Layouts per CDNA5 ISA 7.12.2 (wave32):
//  A 16x32 bf16: lane l -> M=l%16 ; j<8 : K=(l/16)*8+j ; j>=8 : K=16+(l/16)*8+(j-8)
//  B 32x16 bf16 (stored N-major in LDS): lane l -> N=l%16 ; j<8 : K=(l/16)*8+j ; j>=8 : K=16+(l/16)*8+(j-8)
__device__ __forceinline__ void wmma_compute(
    const __bf16* __restrict__ As, const __bf16* __restrict__ Bs,
    int lane, int wm, int wn, v8f acc[2][4]) {
  v16bf afrag[2], bfrag[4];
  const int ml = lane & 15;
  const int h8 = (lane >> 4) * 8;
  const int ks = (lane >> 4) * 16;
#pragma unroll
  for (int mt = 0; mt < 2; ++mt) {
    const __bf16* p = &As[(wm + mt * 16 + ml) * LDT];
#pragma unroll
    for (int j = 0; j < 8; ++j) {
      afrag[mt][j]     = p[h8 + j];
      afrag[mt][j + 8] = p[16 + h8 + j];
    }
  }
#pragma unroll
  for (int nt = 0; nt < 4; ++nt) {
    const __bf16* p = &Bs[(wn + nt * 16 + ml) * LDT];
#pragma unroll
    for (int j = 0; j < 8; ++j) {
      bfrag[nt][j]     = p[h8 + j];
      bfrag[nt][j + 8] = p[16 + h8 + j];
    }
  }
#pragma unroll
  for (int mt = 0; mt < 2; ++mt)
#pragma unroll
    for (int nt = 0; nt < 4; ++nt)
      acc[mt][nt] = __builtin_amdgcn_wmma_f32_16x16x32_bf16(
          false, afrag[mt], false, bfrag[nt], (short)0, acc[mt][nt], false, false);
}

// ---------------------------------------------------------------------------
// GEMM1: Hmid[e] = relu(A[e] @ W1[e]) ; A: [CAP][H] bf16, W1: [H][I] f32
__global__ __launch_bounds__(256) void moe_gemm1(
    const __bf16* __restrict__ A, const float* __restrict__ W1,
    __bf16* __restrict__ Hmid) {
  const int e  = blockIdx.z;
  const int m0 = blockIdx.y * BM;
  const int n0 = blockIdx.x * BN;
  const __bf16* Ae = A    + (size_t)e * CAP_ * H_;
  const float*  We = W1   + (size_t)e * H_ * I_;
  __bf16*       He = Hmid + (size_t)e * CAP_ * I_;

  __shared__ __bf16 As[2][BM * LDT];
  __shared__ __bf16 Bs[2][BN * LDT];

  const int tid  = threadIdx.x;
  const int lane = tid & 31;
  const int wave = tid >> 5;
  const int wm   = (wave >> 2) * 32;
  const int wn   = (wave & 3) * 64;
  const int bn4  = (tid & 63) * 4;     // B-stage: n base
  const int bkr  = tid >> 6;           //          k sub-row 0..3
#if !MOE_HAVE_TDM
  const int ar   = tid >> 2;           // A-stage fallback: row 0..63
  const int ak   = (tid & 3) * 8;      //                   k {0,8,16,24}
  uint4 a_stage;
#endif

  v8f acc[2][4];
#pragma unroll
  for (int i = 0; i < 2; ++i)
#pragma unroll
    for (int j = 0; j < 4; ++j)
#pragma unroll
      for (int r = 0; r < 8; ++r) acc[i][j][r] = 0.f;

  constexpr int NK = H_ / BK;          // 32 K-steps
  float4 b_stage[8];

  auto stage_load = [&](int ksi, int buf) {
    const int k0 = ksi * BK;
#if MOE_HAVE_TDM
    if (wave == 0)                     // one TDM DMA per workgroup per K-step
      tdm_load_a_tile(Ae + (size_t)m0 * H_ + k0,
                      (unsigned)(uintptr_t)&As[buf][0], H_);
#else
    a_stage = *reinterpret_cast<const uint4*>(Ae + (size_t)(m0 + ar) * H_ + k0 + ak);
#endif
#pragma unroll
    for (int p = 0; p < 8; ++p) {
      const int k = p * 4 + bkr;
      b_stage[p] = *reinterpret_cast<const float4*>(We + (size_t)(k0 + k) * I_ + n0 + bn4);
    }
    if (k0 + 2 * BK < H_)              // stream next-next weight tile into L2
      __builtin_prefetch(We + (size_t)(k0 + 2 * BK + bkr) * I_ + n0 + bn4, 0, 1);
  };
  auto stage_store = [&](int buf) {
#if !MOE_HAVE_TDM
    uint32_t* aw = reinterpret_cast<uint32_t*>(&As[buf][ar * LDT + ak]);
    aw[0] = a_stage.x; aw[1] = a_stage.y; aw[2] = a_stage.z; aw[3] = a_stage.w;
#endif
#pragma unroll
    for (int p = 0; p < 8; ++p) {
      const int k = p * 4 + bkr;
      Bs[buf][(bn4 + 0) * LDT + k] = (__bf16)b_stage[p].x;
      Bs[buf][(bn4 + 1) * LDT + k] = (__bf16)b_stage[p].y;
      Bs[buf][(bn4 + 2) * LDT + k] = (__bf16)b_stage[p].z;
      Bs[buf][(bn4 + 3) * LDT + k] = (__bf16)b_stage[p].w;
    }
  };

  stage_load(0, 0);
  stage_store(0);
#if MOE_HAVE_TDM
  if (wave == 0) __builtin_amdgcn_s_wait_tensorcnt(0);
#endif
  __syncthreads();
  for (int ksi = 0; ksi < NK; ++ksi) {
    const int cur = ksi & 1;
    if (ksi + 1 < NK) stage_load(ksi + 1, cur ^ 1);
    wmma_compute(As[cur], Bs[cur], lane, wm, wn, acc);
    __syncthreads();
    if (ksi + 1 < NK) {
      stage_store(cur ^ 1);
#if MOE_HAVE_TDM
      if (wave == 0) __builtin_amdgcn_s_wait_tensorcnt(0);
#endif
      __syncthreads();
    }
  }

  // Epilogue: relu -> bf16 store. C layout: lane -> N=l%16, VGPR r -> M=8*(l/16)+r
  const int ml = lane & 15;
  const int mh = (lane >> 4) * 8;
#pragma unroll
  for (int mt = 0; mt < 2; ++mt)
#pragma unroll
    for (int nt = 0; nt < 4; ++nt) {
      const int col = n0 + wn + nt * 16 + ml;
#pragma unroll
      for (int r = 0; r < 8; ++r) {
        const int row = m0 + wm + mt * 16 + mh + r;
        float v = acc[mt][nt][r];
        v = v > 0.f ? v : 0.f;
        He[(size_t)row * I_ + col] = (__bf16)v;
      }
    }
}

// ---------------------------------------------------------------------------
// GEMM2: out += weight[slot] * (Hmid[e] @ W2[e]) scattered to token rows
__global__ __launch_bounds__(256) void moe_gemm2(
    const __bf16* __restrict__ Hmid, const float* __restrict__ W2,
    const int* __restrict__ slot_token, const float* __restrict__ slot_weight,
    float* __restrict__ out) {
  const int e  = blockIdx.z;
  const int m0 = blockIdx.y * BM;
  const int n0 = blockIdx.x * BN;
  const __bf16* Ae = Hmid + (size_t)e * CAP_ * I_;
  const float*  We = W2   + (size_t)e * I_ * H_;

  __shared__ __bf16 As[2][BM * LDT];
  __shared__ __bf16 Bs[2][BN * LDT];

  const int tid  = threadIdx.x;
  const int lane = tid & 31;
  const int wave = tid >> 5;
  const int wm   = (wave >> 2) * 32;
  const int wn   = (wave & 3) * 64;
  const int bn4  = (tid & 63) * 4;
  const int bkr  = tid >> 6;
#if !MOE_HAVE_TDM
  const int ar   = tid >> 2;
  const int ak   = (tid & 3) * 8;
  uint4 a_stage;
#endif

  v8f acc[2][4];
#pragma unroll
  for (int i = 0; i < 2; ++i)
#pragma unroll
    for (int j = 0; j < 4; ++j)
#pragma unroll
      for (int r = 0; r < 8; ++r) acc[i][j][r] = 0.f;

  constexpr int NK = I_ / BK;          // 128 K-steps
  float4 b_stage[8];

  auto stage_load = [&](int ksi, int buf) {
    const int k0 = ksi * BK;
#if MOE_HAVE_TDM
    if (wave == 0)
      tdm_load_a_tile(Ae + (size_t)m0 * I_ + k0,
                      (unsigned)(uintptr_t)&As[buf][0], I_);
#else
    a_stage = *reinterpret_cast<const uint4*>(Ae + (size_t)(m0 + ar) * I_ + k0 + ak);
#endif
#pragma unroll
    for (int p = 0; p < 8; ++p) {
      const int k = p * 4 + bkr;
      b_stage[p] = *reinterpret_cast<const float4*>(We + (size_t)(k0 + k) * H_ + n0 + bn4);
    }
    if (k0 + 2 * BK < I_)
      __builtin_prefetch(We + (size_t)(k0 + 2 * BK + bkr) * H_ + n0 + bn4, 0, 1);
  };
  auto stage_store = [&](int buf) {
#if !MOE_HAVE_TDM
    uint32_t* aw = reinterpret_cast<uint32_t*>(&As[buf][ar * LDT + ak]);
    aw[0] = a_stage.x; aw[1] = a_stage.y; aw[2] = a_stage.z; aw[3] = a_stage.w;
#endif
#pragma unroll
    for (int p = 0; p < 8; ++p) {
      const int k = p * 4 + bkr;
      Bs[buf][(bn4 + 0) * LDT + k] = (__bf16)b_stage[p].x;
      Bs[buf][(bn4 + 1) * LDT + k] = (__bf16)b_stage[p].y;
      Bs[buf][(bn4 + 2) * LDT + k] = (__bf16)b_stage[p].z;
      Bs[buf][(bn4 + 3) * LDT + k] = (__bf16)b_stage[p].w;
    }
  };

  stage_load(0, 0);
  stage_store(0);
#if MOE_HAVE_TDM
  if (wave == 0) __builtin_amdgcn_s_wait_tensorcnt(0);
#endif
  __syncthreads();
  for (int ksi = 0; ksi < NK; ++ksi) {
    const int cur = ksi & 1;
    if (ksi + 1 < NK) stage_load(ksi + 1, cur ^ 1);
    wmma_compute(As[cur], Bs[cur], lane, wm, wn, acc);
    __syncthreads();
    if (ksi + 1 < NK) {
      stage_store(cur ^ 1);
#if MOE_HAVE_TDM
      if (wave == 0) __builtin_amdgcn_s_wait_tensorcnt(0);
#endif
      __syncthreads();
    }
  }

  // Epilogue: scale by routing weight, scatter-add to token rows.
  const int ml = lane & 15;
  const int mh = (lane >> 4) * 8;
#pragma unroll
  for (int mt = 0; mt < 2; ++mt)
#pragma unroll
    for (int nt = 0; nt < 4; ++nt) {
      const int col = n0 + wn + nt * 16 + ml;
#pragma unroll
      for (int r = 0; r < 8; ++r) {
        const int row = m0 + wm + mt * 16 + mh + r;
        const int token = slot_token[e * CAP_ + row];
        if (token >= 0) {
          const float w = slot_weight[e * CAP_ + row];
          __hip_atomic_fetch_add(&out[(size_t)token * H_ + col],
                                 acc[mt][nt][r] * w,
                                 __ATOMIC_RELAXED, __HIP_MEMORY_SCOPE_AGENT);
        }
      }
    }
}

// ===========================================================================
extern "C" void kernel_launch(void* const* d_in, const int* in_sizes, int n_in,
                              void* d_out, int out_size, void* d_ws, size_t ws_size,
                              hipStream_t stream) {
  const float* x  = (const float*)d_in[0];   // [T, H]
  const float* Wr = (const float*)d_in[1];   // [E, H]
  const float* br = (const float*)d_in[2];   // [E]
  const float* W1 = (const float*)d_in[3];   // [E, H, I]
  const float* W2 = (const float*)d_in[4];   // [E, I, H]
  float* out = (float*)d_out;                // [T, H]

  // workspace layout (~105 MB)
  char* ws = (char*)d_ws;
  float*  topk_w      = (float*)(ws);                       // T*E f32   (256 KB)
  int*    slot_token  = (int*)  (ws + 262144);              // E*CAP i32 (40 KB)
  float*  slot_weight = (float*)(ws + 303104);              // E*CAP f32 (40 KB)
  __bf16* A_bf16      = (__bf16*)(ws + 344064);             // E*CAP*H   (20 MB)
  __bf16* Hmid        = (__bf16*)(ws + 21315584);           // E*CAP*I   (80 MB)

  moe_zero<<<(T_ * H_ / 4) / 256, 256, 0, stream>>>(out);
  moe_router<<<T_, 256, 0, stream>>>(x, Wr, br, topk_w);
  moe_assign<<<E_, 1024, 0, stream>>>(topk_w, slot_token, slot_weight);
  moe_gather<<<dim3(CAP_, E_), 256, 0, stream>>>(x, slot_token, A_bf16);
  moe_gemm1<<<dim3(I_ / BN, CAP_ / BM, E_), 256, 0, stream>>>(A_bf16, W1, Hmid);
  moe_gemm2<<<dim3(H_ / BN, CAP_ / BM, E_), 256, 0, stream>>>(Hmid, W2, slot_token,
                                                              slot_weight, out);
}
